// SparseMatchingStereoModel_4277787427054
// MI455X (gfx1250) — matrix operand order repaired and checked by reading, not compiled, not run on hardware
//
#include <hip/hip_runtime.h>
#include <hip/hip_bf16.h>

typedef __attribute__((ext_vector_type(16))) _Float16 v16h;
typedef __attribute__((ext_vector_type(8)))  _Float16 v8h;
typedef __attribute__((ext_vector_type(4)))  _Float16 v4h;
typedef __attribute__((ext_vector_type(8)))  float    v8f;

#define BB 8
#define NN 1024
#define DD 768
#define LL 6
#define NHD 8
#define DHC 96

// Fragment loaders: two contiguous 16B LDS reads -> ds_load_b128.
// A-matrix fragment: K = g*16 .. g*16+15 contiguous (matches frag_B)
__device__ __forceinline__ v16h frag_A(const _Float16* base, int g) {
    v8h lo = *(const v8h*)(base + g * 16);
    v8h hi = *(const v8h*)(base + g * 16 + 8);
    return __builtin_shufflevector(lo, hi, 0, 1, 2, 3, 4, 5, 6, 7,
                                           8, 9, 10, 11, 12, 13, 14, 15);
}
// B-matrix fragment: K = g*16 .. g*16+15 contiguous (tile stored N-major, K-contig)
__device__ __forceinline__ v16h frag_B(const _Float16* base, int g) {
    v8h lo = *(const v8h*)(base + g * 16);
    v8h hi = *(const v8h*)(base + g * 16 + 8);
    return __builtin_shufflevector(lo, hi, 0, 1, 2, 3, 4, 5, 6, 7,
                                           8, 9, 10, 11, 12, 13, 14, 15);
}

__device__ __forceinline__ v4h cvt4(const float4 f) {
    v4h h; h[0] = (_Float16)f.x; h[1] = (_Float16)f.y;
    h[2] = (_Float16)f.z; h[3] = (_Float16)f.w; return h;
}

// ---------------------------------------------------------------------------
// GEMM: C[M,Nc] = act(A[M,K] @ W[Nc,K]^T + bias) (+ residual), batched by z.
// 128x64 macro tile, K-step 64, 8 waves, wave = 32x32 (2x2 WMMA tiles, 8
// WMMA / stage).  Software pipelined: next tile's global loads issue before
// the current tile's WMMA phase.
// ---------------------------------------------------------------------------
__global__ __launch_bounds__(256)
void gemm_wmma_kernel(const float* __restrict__ A, const float* __restrict__ Wt,
                      const float* __restrict__ bias, const float* res,
                      float* C, int M, int Nc, int K,
                      long long sA, long long sW, long long sC, int relu)
{
    __shared__ __align__(16) _Float16 As[128 * 64];  // [m][k]
    __shared__ __align__(16) _Float16 Bs[64 * 64];   // [n][k]

    const long long bz = blockIdx.z;
    A  += bz * sA;
    Wt += bz * sW;
    C  += bz * sC;
    if (res) res += bz * sC;

    const int n0 = blockIdx.x * 64;
    const int m0 = blockIdx.y * 128;
    const int t    = threadIdx.x;
    const int w    = t >> 5;
    const int lane = t & 31;
    const int g    = lane >> 4;
    const int mr   = lane & 15;
    const int wr   = w & 3;       // wave row (32 rows each)
    const int wc   = w >> 2;      // wave col (32 cols each)

    v8f acc[2][2] = {};
    const int nsteps = K >> 6;

    float4 ra[8], rb[4];
    // prologue: stage tile 0 into registers (clause of b128 loads)
    #pragma unroll
    for (int i = 0; i < 8; ++i) {
        int flat = (i * 256 + t) * 4;
        int r = flat >> 6, c = flat & 63;
        ra[i] = *(const float4*)&A[(long long)(m0 + r) * K + c];
    }
    #pragma unroll
    for (int i = 0; i < 4; ++i) {
        int flat = (i * 256 + t) * 4;
        int r = flat >> 6, c = flat & 63;
        rb[i] = *(const float4*)&Wt[(long long)(n0 + r) * K + c];
    }

    for (int ks = 0; ks < nsteps; ++ks) {
        // registers -> LDS (f32 -> f16 packed)
        #pragma unroll
        for (int i = 0; i < 8; ++i) {
            int flat = (i * 256 + t) * 4;
            int r = flat >> 6, c = flat & 63;
            *(v4h*)&As[r * 64 + c] = cvt4(ra[i]);
        }
        #pragma unroll
        for (int i = 0; i < 4; ++i) {
            int flat = (i * 256 + t) * 4;
            int r = flat >> 6, c = flat & 63;
            *(v4h*)&Bs[r * 64 + c] = cvt4(rb[i]);
        }
        __syncthreads();

        // issue next tile's global loads; consumed only at next iteration
        if (ks + 1 < nsteps) {
            const int k0 = (ks + 1) << 6;
            #pragma unroll
            for (int i = 0; i < 8; ++i) {
                int flat = (i * 256 + t) * 4;
                int r = flat >> 6, c = flat & 63;
                ra[i] = *(const float4*)&A[(long long)(m0 + r) * K + (k0 + c)];
            }
            #pragma unroll
            for (int i = 0; i < 4; ++i) {
                int flat = (i * 256 + t) * 4;
                int r = flat >> 6, c = flat & 63;
                rb[i] = *(const float4*)&Wt[(long long)(n0 + r) * K + (k0 + c)];
            }
        }

        // WMMA phase: 2 k-steps x 2x2 tiles
        #pragma unroll
        for (int kk = 0; kk < 64; kk += 32) {
            v16h af0 = frag_A(&As[(wr * 32 + mr) * 64 + kk], g);
            v16h af1 = frag_A(&As[(wr * 32 + 16 + mr) * 64 + kk], g);
            v16h bf0 = frag_B(&Bs[(wc * 32 + mr) * 64 + kk], g);
            v16h bf1 = frag_B(&Bs[(wc * 32 + 16 + mr) * 64 + kk], g);
            acc[0][0] = __builtin_amdgcn_wmma_f32_16x16x32_f16(false, af0, false, bf0,
                                                               (short)0, acc[0][0], false, false);
            acc[0][1] = __builtin_amdgcn_wmma_f32_16x16x32_f16(false, af0, false, bf1,
                                                               (short)0, acc[0][1], false, false);
            acc[1][0] = __builtin_amdgcn_wmma_f32_16x16x32_f16(false, af1, false, bf0,
                                                               (short)0, acc[1][0], false, false);
            acc[1][1] = __builtin_amdgcn_wmma_f32_16x16x32_f16(false, af1, false, bf1,
                                                               (short)0, acc[1][1], false, false);
        }
        __syncthreads();
    }

    #pragma unroll
    for (int i2 = 0; i2 < 2; ++i2) {
        #pragma unroll
        for (int j2 = 0; j2 < 2; ++j2) {
            #pragma unroll
            for (int i = 0; i < 8; ++i) {
                int row = m0 + wr * 32 + i2 * 16 + g * 8 + i;
                int col = n0 + wc * 32 + j2 * 16 + mr;
                float v = acc[i2][j2][i];
                if (bias) v += bias[col];
                if (relu) v = fmaxf(v, 0.f);
                if (res)  v += res[(long long)row * Nc + col];
                C[(long long)row * Nc + col] = v;
            }
        }
    }
}

// ---------------------------------------------------------------------------
// Fused flash attention, dh = 96, 64-query tile per block.
// grid = (N/64, NH, B), block = 256 (8 waves).  Next key-tile's K/V global
// loads issue before the current tile's compute phases.
// ---------------------------------------------------------------------------
__global__ __launch_bounds__(256)
void attn_kernel(const float* __restrict__ Q, const float* __restrict__ Kf,
                 const float* __restrict__ Vf, float* __restrict__ O,
                 int Nq, int Nk, int Dtot)
{
    __shared__ __align__(16) _Float16 Qs[64 * DHC];  // [q][d]
    __shared__ __align__(16) _Float16 Ks[64 * DHC];  // [k][d]
    __shared__ __align__(16) _Float16 Vt[DHC * 64];  // [d][k]  (transposed)
    __shared__ __align__(16) float    Ss[64 * 64];
    __shared__ __align__(16) _Float16 Ps[64 * 64];   // [q][k]
    __shared__ float pred[256];
    __shared__ float rowm[64], rowl[64], rowsc[64];

    const int qb = blockIdx.x * 64;
    const int h  = blockIdx.y;
    const int b  = blockIdx.z;
    const long long baseQ = (long long)b * Nq * Dtot + h * DHC;
    const long long baseK = (long long)b * Nk * Dtot + h * DHC;
    const int t = threadIdx.x, w = t >> 5, lane = t & 31;
    const int g = lane >> 4, mr = lane & 15;
    const int sr = t & 63, ch = t >> 6;     // softmax row / column chunk
    const float scale = 0.1020620726159658f;  // 1/sqrt(96)

    // stage Q (hoisted loads -> one clause, then convert/store)
    {
        float4 rq[6];
        #pragma unroll
        for (int i = 0; i < 6; ++i) {
            int flat = (i * 256 + t) * 4;
            int r = flat / DHC, c = flat % DHC;
            rq[i] = *(const float4*)&Q[baseQ + (long long)(qb + r) * Dtot + c];
        }
        #pragma unroll
        for (int i = 0; i < 6; ++i) {
            int flat = (i * 256 + t) * 4;
            int r = flat / DHC, c = flat % DHC;
            *(v4h*)&Qs[r * DHC + c] = cvt4(rq[i]);
        }
    }
    if (t < 64) { rowm[t] = -3.402823466e38f; rowl[t] = 0.f; }

    v8f oacc[3] = {};
    float4 rk[6], rv[6];
    #pragma unroll
    for (int i = 0; i < 6; ++i) {
        int flat = (i * 256 + t) * 4;
        int r = flat / DHC, c = flat % DHC;
        rk[i] = *(const float4*)&Kf[baseK + (long long)r * Dtot + c];
        rv[i] = *(const float4*)&Vf[baseK + (long long)r * Dtot + c];
    }
    __syncthreads();

    for (int kt = 0; kt < Nk; kt += 64) {
        // staged registers -> LDS
        #pragma unroll
        for (int i = 0; i < 6; ++i) {
            int flat = (i * 256 + t) * 4;
            int r = flat / DHC, c = flat % DHC;
            *(v4h*)&Ks[r * DHC + c] = cvt4(rk[i]);
            v4h hv = cvt4(rv[i]);
            #pragma unroll
            for (int j = 0; j < 4; ++j) Vt[(c + j) * 64 + r] = hv[j];
        }
        __syncthreads();

        // issue next key tile's loads; consumed next iteration
        if (kt + 64 < Nk) {
            #pragma unroll
            for (int i = 0; i < 6; ++i) {
                int flat = (i * 256 + t) * 4;
                int r = flat / DHC, c = flat % DHC;
                rk[i] = *(const float4*)&Kf[baseK + (long long)(kt + 64 + r) * Dtot + c];
                rv[i] = *(const float4*)&Vf[baseK + (long long)(kt + 64 + r) * Dtot + c];
            }
        }

        // S = (Q K^T) * scale ; wave computes tiles 2w and 2w+1 of 4x4 grid
        #pragma unroll
        for (int j = 0; j < 2; ++j) {
            const int tid = 2 * w + j;
            const int str = tid >> 2, stc = tid & 3;
            v8f sa = {};
            #pragma unroll
            for (int kk = 0; kk < DHC; kk += 32) {
                v16h af = frag_A(&Qs[(str * 16 + mr) * DHC + kk], g);
                v16h bf = frag_B(&Ks[(stc * 16 + mr) * DHC + kk], g);
                sa = __builtin_amdgcn_wmma_f32_16x16x32_f16(false, af, false, bf,
                                                            (short)0, sa, false, false);
            }
            #pragma unroll
            for (int i = 0; i < 8; ++i)
                Ss[(str * 16 + g * 8 + i) * 64 + (stc * 16 + mr)] = sa[i] * scale;
        }
        __syncthreads();

        // online softmax, 4 threads per row
        {
            float mx = -3.402823466e38f;
            #pragma unroll
            for (int c = 0; c < 16; ++c) mx = fmaxf(mx, Ss[sr * 64 + ch * 16 + c]);
            pred[ch * 64 + sr] = mx;
        }
        __syncthreads();
        if (t < 64) {
            float om = rowm[t];
            float m4 = fmaxf(fmaxf(pred[t], pred[64 + t]),
                             fmaxf(pred[128 + t], pred[192 + t]));
            float nm = fmaxf(om, m4);
            rowm[t] = nm;
            rowsc[t] = expf(om - nm);
        }
        __syncthreads();
        {
            float nm = rowm[sr];
            float sum = 0.f;
            #pragma unroll
            for (int c = 0; c < 16; ++c) {
                float p = expf(Ss[sr * 64 + ch * 16 + c] - nm);
                Ps[sr * 64 + ch * 16 + c] = (_Float16)p;
                sum += p;
            }
            pred[ch * 64 + sr] = sum;
        }
        __syncthreads();
        if (t < 64)
            rowl[t] = rowl[t] * rowsc[t] +
                      (pred[t] + pred[64 + t]) + (pred[128 + t] + pred[192 + t]);
        __syncthreads();

        // rescale O accumulators, then O += P * V (wave owns 3 of 4x6 tiles)
        #pragma unroll
        for (int j = 0; j < 3; ++j) {
            const int tid = 3 * w + j;
            const int otr = tid / 6, otc = tid % 6;
            v8f a = oacc[j];
            #pragma unroll
            for (int i = 0; i < 8; ++i) a[i] *= rowsc[otr * 16 + g * 8 + i];
            #pragma unroll
            for (int kk = 0; kk < 64; kk += 32) {
                v16h af = frag_A(&Ps[(otr * 16 + mr) * 64 + kk], g);
                v16h bf = frag_B(&Vt[(otc * 16 + mr) * 64 + kk], g);
                a = __builtin_amdgcn_wmma_f32_16x16x32_f16(false, af, false, bf,
                                                           (short)0, a, false, false);
            }
            oacc[j] = a;
        }
        __syncthreads();
    }

    #pragma unroll
    for (int j = 0; j < 3; ++j) {
        const int tid = 3 * w + j;
        const int otr = tid / 6, otc = tid % 6;
        #pragma unroll
        for (int i = 0; i < 8; ++i) {
            int r = otr * 16 + g * 8 + i;
            float l = rowl[r];
            float inv = (l > 0.f) ? 1.f / l : 0.f;
            O[baseQ + (long long)(qb + r) * Dtot + otc * 16 + mr] = oacc[j][i] * inv;
        }
    }
}

// ---------------------------------------------------------------------------
// Row LayerNorm over D=768 (optionally adds `addp` first). 1 block / row.
// ---------------------------------------------------------------------------
__global__ __launch_bounds__(256)
void ln_kernel(const float* __restrict__ X, const float* __restrict__ addp,
               const float* __restrict__ gamma, const float* __restrict__ beta,
               float* __restrict__ Y)
{
    __shared__ float red[256];
    const long long row = blockIdx.x;
    const int t = threadIdx.x;
    const float* x = X + row * DD;
    float a0 = x[t], a1 = x[t + 256], a2 = x[t + 512];
    if (addp) {
        const float* p = addp + row * DD;
        a0 += p[t]; a1 += p[t + 256]; a2 += p[t + 512];
    }
    red[t] = a0 + a1 + a2;
    __syncthreads();
    for (int s = 128; s > 0; s >>= 1) { if (t < s) red[t] += red[t + s]; __syncthreads(); }
    float mean = red[0] * (1.0f / DD);
    __syncthreads();
    float d0 = a0 - mean, d1 = a1 - mean, d2 = a2 - mean;
    red[t] = d0 * d0 + d1 * d1 + d2 * d2;
    __syncthreads();
    for (int s = 128; s > 0; s >>= 1) { if (t < s) red[t] += red[t + s]; __syncthreads(); }
    float var = red[0] * (1.0f / DD);
    float r = rsqrtf(var + 1e-5f);
    float* y = Y + row * DD;
    y[t]       = d0 * r * gamma[t]       + beta[t];
    y[t + 256] = d1 * r * gamma[t + 256] + beta[t + 256];
    y[t + 512] = d2 * r * gamma[t + 512] + beta[t + 512];
}

// ---------------------------------------------------------------------------
// Row L2 normalize (F.normalize, eps=1e-8). 1 block / row.
// ---------------------------------------------------------------------------
__global__ __launch_bounds__(256)
void l2norm_kernel(const float* __restrict__ X, float* __restrict__ Y)
{
    __shared__ float red[256];
    const long long row = blockIdx.x;
    const int t = threadIdx.x;
    const float* x = X + row * DD;
    float a0 = x[t], a1 = x[t + 256], a2 = x[t + 512];
    red[t] = a0 * a0 + a1 * a1 + a2 * a2;
    __syncthreads();
    for (int s = 128; s > 0; s >>= 1) { if (t < s) red[t] += red[t + s]; __syncthreads(); }
    float inv = 1.0f / fmaxf(sqrtf(red[0]), 1e-8f);
    float* y = Y + row * DD;
    y[t] = a0 * inv; y[t + 256] = a1 * inv; y[t + 512] = a2 * inv;
}

// ---------------------------------------------------------------------------
// Positional encoding: pos = (kp * [1/(W-1), 1/(H-1)]) @ pos_w^T + pos_b
// ---------------------------------------------------------------------------
__global__ __launch_bounds__(256)
void pos_kernel(const float* __restrict__ kp, const float* __restrict__ pw,
                const float* __restrict__ pb, const int* __restrict__ Wp,
                const int* __restrict__ Hp, float* __restrict__ out)
{
    long long idx = (long long)blockIdx.x * 256 + threadIdx.x;
    const long long total = (long long)BB * NN * DD;
    if (idx >= total) return;
    int d = (int)(idx % DD);
    long long bn = idx / DD;
    float invw = 1.0f / (float)(Wp[0] - 1);
    float invh = 1.0f / (float)(Hp[0] - 1);
    float2 kpv = *(const float2*)&kp[bn * 2];
    float xs = kpv.x * invw;
    float ys = kpv.y * invh;
    out[idx] = xs * pw[d * 2] + ys * pw[d * 2 + 1] + pb[d];
}

// ---------------------------------------------------------------------------
// Masked softmax over keys + soft disparity. grid = (N, B), block = 256.
// ---------------------------------------------------------------------------
__global__ __launch_bounds__(256)
void match_kernel(const float* __restrict__ S, const float* __restrict__ kpl,
                  const float* __restrict__ kpr, float* __restrict__ probs,
                  float* __restrict__ disp)
{
    __shared__ float red[256];
    const int n = blockIdx.x, b = blockIdx.y, t = threadIdx.x;
    const long long rowi = (long long)b * NN + n;
    const float* srow = S + rowi * NN;
    const float2 kl = *(const float2*)&kpl[rowi * 2];
    const float xl = kl.x, yl = kl.y;
    const float2* kr2 = (const float2*)(kpr + (long long)b * NN * 2);

    float lg[4], xr4[4];
    float mx = -3.402823466e38f;
    #pragma unroll
    for (int j = 0; j < 4; ++j) {
        int m = t + j * 256;
        float2 kv = kr2[m];
        bool ok = (xl >= kv.x) && (fabsf(yl - kv.y) < 2.0f);
        float v = ok ? srow[m] * 10.0f : -3.402823466e38f;
        lg[j] = v; xr4[j] = kv.x;
        mx = fmaxf(mx, v);
    }
    red[t] = mx; __syncthreads();
    for (int s = 128; s > 0; s >>= 1) { if (t < s) red[t] = fmaxf(red[t], red[t + s]); __syncthreads(); }
    mx = red[0]; __syncthreads();

    float p4[4], sum = 0.f, dsum = 0.f;
    #pragma unroll
    for (int j = 0; j < 4; ++j) {
        float p = expf(lg[j] - mx);
        p4[j] = p; sum += p; dsum += (xl - xr4[j]) * p;
    }
    red[t] = sum; __syncthreads();
    for (int s = 128; s > 0; s >>= 1) { if (t < s) red[t] += red[t + s]; __syncthreads(); }
    sum = red[0]; __syncthreads();
    red[t] = dsum; __syncthreads();
    for (int s = 128; s > 0; s >>= 1) { if (t < s) red[t] += red[t + s]; __syncthreads(); }
    dsum = red[0];

    float inv = (sum > 0.f) ? 1.0f / sum : 0.f;
    #pragma unroll
    for (int j = 0; j < 4; ++j)
        probs[rowi * NN + t + j * 256] = p4[j] * inv;
    if (t == 0) {
        float dv = dsum * inv;
        if (!__builtin_isfinite(dv)) dv = 0.f;
        disp[rowi] = dv;
    }
}

// ---------------------------------------------------------------------------
// Host orchestration
// ---------------------------------------------------------------------------
extern "C" void kernel_launch(void* const* d_in, const int* in_sizes, int n_in,
                              void* d_out, int out_size, void* d_ws, size_t ws_size,
                              hipStream_t stream)
{
    (void)in_sizes; (void)n_in; (void)out_size; (void)ws_size;

    const float* desc_l = (const float*)d_in[0];
    const float* desc_r = (const float*)d_in[1];
    const float* kp_l   = (const float*)d_in[2];
    const float* kp_r   = (const float*)d_in[3];
    const float* pos_w  = (const float*)d_in[4];
    const float* pos_b  = (const float*)d_in[5];
    // params_left: 6..21  params_right: 22..37  cross: 38..51  H:52  W:53
    enum { WQ = 0, WK, WV, WO, BQi, BKi, BVi, BOi, LN1G, LN1B, LN2G, LN2B, W1, B1, W2, B2 };
    const float* ppl[16]; const float* ppr[16];
    for (int j = 0; j < 16; ++j) { ppl[j] = (const float*)d_in[6 + j]; ppr[j] = (const float*)d_in[22 + j]; }
    const float* cx[14];
    for (int j = 0; j < 14; ++j) cx[j] = (const float*)d_in[38 + j];
    enum { CWQ = 0, CWK, CWV, CWO, CBQ, CBK, CBV, CBO, NQG, NQB, NKVG, NKVB, NAG, NAB };
    const int* Hp = (const int*)d_in[52];
    const int* Wp = (const int*)d_in[53];

    const long long SZ = (long long)BB * NN * DD;      // 6,291,456
    float* ws   = (float*)d_ws;
    float* posl = ws;
    float* posr = posl + SZ;
    float* fl   = posr + SZ;
    float* fr   = fl + SZ;
    float* xb   = fr + SZ;
    float* qb   = xb + SZ;
    float* kb   = qb + SZ;
    float* vb   = kb + SZ;
    float* ob   = vb + SZ;
    float* hb   = ob + SZ;          // 2*SZ (FFN hidden); reused for scores
    float* scores = hb;             // B*N*N = 8.39M < 2*SZ = 12.58M

    const int M = BB * NN;          // 8192

    auto gemm = [&](const float* A, const float* Wm, const float* bias,
                    const float* res, float* C, int Mm, int Nc, int K,
                    int relu, int batch, long long sA, long long sW, long long sC) {
        dim3 grid(Nc / 64, Mm / 128, batch);
        gemm_wmma_kernel<<<grid, 256, 0, stream>>>(A, Wm, bias, res, C, Mm, Nc, K,
                                                   sA, sW, sC, relu);
    };
    auto ln = [&](const float* X, const float* addp, const float* g,
                  const float* bta, float* Y) {
        ln_kernel<<<dim3(M), 256, 0, stream>>>(X, addp, g, bta, Y);
    };

    // positional encodings
    {
        long long total = SZ;
        int nb = (int)((total + 255) / 256);
        pos_kernel<<<dim3(nb), 256, 0, stream>>>(kp_l, pos_w, pos_b, Wp, Hp, posl);
        pos_kernel<<<dim3(nb), 256, 0, stream>>>(kp_r, pos_w, pos_b, Wp, Hp, posr);
    }
    hipMemcpyAsync(fl, desc_l, SZ * sizeof(float), hipMemcpyDeviceToDevice, stream);
    hipMemcpyAsync(fr, desc_r, SZ * sizeof(float), hipMemcpyDeviceToDevice, stream);

    auto run_side = [&](float* f, const float* pos, const float* const* pp) {
        for (int i = 0; i < LL; ++i) {
            const long long oW = (long long)i * DD * DD;
            const long long oB = (long long)i * DD;
            const long long oW1 = (long long)i * 2 * DD * DD;
            const long long oB1 = (long long)i * 2 * DD;
            // x = LN(f + pos)
            ln(f, pos, pp[LN1G] + oB, pp[LN1B] + oB, xb);
            // q, k, v projections
            gemm(xb, pp[WQ] + oW, pp[BQi] + oB, nullptr, qb, M, DD, DD, 0, 1, 0, 0, 0);
            gemm(xb, pp[WK] + oW, pp[BKi] + oB, nullptr, kb, M, DD, DD, 0, 1, 0, 0, 0);
            gemm(xb, pp[WV] + oW, pp[BVi] + oB, nullptr, vb, M, DD, DD, 0, 1, 0, 0, 0);
            // fused attention
            attn_kernel<<<dim3(NN / 64, NHD, BB), 256, 0, stream>>>(qb, kb, vb, ob, NN, NN, DD);
            // f = f + o @ Wo^T + bo
            gemm(ob, pp[WO] + oW, pp[BOi] + oB, f, f, M, DD, DD, 0, 1, 0, 0, 0);
            // FFN
            ln(f, nullptr, pp[LN2G] + oB, pp[LN2B] + oB, xb);
            gemm(xb, pp[W1] + oW1, pp[B1] + oB1, nullptr, hb, M, 2 * DD, DD, 1, 1, 0, 0, 0);
            gemm(hb, pp[W2] + oW1, pp[B2] + oB, f, f, M, DD, 2 * DD, 0, 1, 0, 0, 0);
        }
    };
    run_side(fl, posl, ppl);
    run_side(fr, posr, ppr);

    // cross attention: left queries right
    ln(fl, nullptr, cx[NQG], cx[NQB], xb);
    gemm(xb, cx[CWQ], cx[CBQ], nullptr, qb, M, DD, DD, 0, 1, 0, 0, 0);
    ln(fr, nullptr, cx[NKVG], cx[NKVB], xb);
    gemm(xb, cx[CWK], cx[CBK], nullptr, kb, M, DD, DD, 0, 1, 0, 0, 0);
    gemm(xb, cx[CWV], cx[CBV], nullptr, vb, M, DD, DD, 0, 1, 0, 0, 0);
    attn_kernel<<<dim3(NN / 64, NHD, BB), 256, 0, stream>>>(qb, kb, vb, ob, NN, NN, DD);
    gemm(ob, cx[CWO], cx[CBO], fl, fl, M, DD, DD, 0, 1, 0, 0, 0);
    ln(fl, nullptr, cx[NAG], cx[NAB], xb);    // final fused feature in xb

    // normalized features and score matrix (batched GEMM: scores = fln @ frn^T)
    l2norm_kernel<<<dim3(M), 256, 0, stream>>>(xb, qb);   // fln
    l2norm_kernel<<<dim3(M), 256, 0, stream>>>(fr, kb);   // frn
    gemm(qb, kb, nullptr, nullptr, scores, NN, NN, DD, 0, BB,
         (long long)NN * DD, (long long)NN * DD, (long long)NN * NN);

    // masked softmax + disparity -> outputs
    float* probs = (float*)d_out;
    float* disp  = probs + (long long)BB * NN * NN;
    match_kernel<<<dim3(NN, BB), 256, 0, stream>>>(scores, kp_l, kp_r, probs, disp);
}
